// CustomMultiLatentAttention_18803366822021
// MI455X (gfx1250) — matrix operand order repaired and checked by reading, not compiled, not run on hardware
//
#include <hip/hip_runtime.h>
#include <hip/hip_bf16.h>
#include <math.h>

// ---------------------------------------------------------------------------
// MLA (multi-latent attention) for gfx1250 / MI455X.
// GEMMs + attention einsums via V_WMMA_F32_16X16X32_BF16 (wave32).
// Attention K/V staged through LDS by the Tensor Data Mover (TDM), with
// s_wait_tensorcnt synchronization (CDNA5 async-tensor path).
// ---------------------------------------------------------------------------

typedef __attribute__((ext_vector_type(16))) __bf16 bf16x16;
typedef __attribute__((ext_vector_type(8)))  __bf16 bf16x8;
typedef __attribute__((ext_vector_type(8)))  float  f32x8;
typedef __attribute__((ext_vector_type(4)))  unsigned int v4u;
typedef __attribute__((ext_vector_type(8)))  int v8i;
typedef __attribute__((ext_vector_type(4)))  int v4i;

#define B_   2
#define S_   2048
#define H_   2048
#define NH_  16
#define DQK_ 128
#define DPE_ 64
#define DV_  128
#define QHD_ 192
#define LAT_ 512

__device__ __forceinline__ f32x8 wmma_bf(bf16x16 a, bf16x16 b, f32x8 c) {
  return __builtin_amdgcn_wmma_f32_16x16x32_bf16(false, a, false, b, (short)0, c,
                                                 false, false);
}

__device__ __forceinline__ bf16x16 cat8(bf16x8 lo, bf16x8 hi) {
  return __builtin_shufflevector(lo, hi, 0, 1, 2, 3, 4, 5, 6, 7, 8, 9, 10, 11,
                                 12, 13, 14, 15);
}

// A-frag (16x32): lane m = lane&15, hi = lane>>4; k in {16hi..16hi+15}
__device__ __forceinline__ bf16x16 lda(const __bf16* row, int kb, int hi) {
  bf16x8 lo = *(const bf16x8*)(row + kb + 16 * hi);
  bf16x8 h8 = *(const bf16x8*)(row + kb + 16 * hi + 8);
  return cat8(lo, h8);
}
// B-frag (32x16): lane n = lane&15; k in {16hi..16hi+15} (one 32-B load)
__device__ __forceinline__ bf16x16 ldb(const __bf16* row, int kb, int hi) {
  return *(const bf16x16*)(row + kb + 16 * hi);
}

__device__ __forceinline__ float rmax16(float v) {
  v = fmaxf(v, __shfl_xor(v, 1, 32));
  v = fmaxf(v, __shfl_xor(v, 2, 32));
  v = fmaxf(v, __shfl_xor(v, 4, 32));
  v = fmaxf(v, __shfl_xor(v, 8, 32));
  return v;
}
__device__ __forceinline__ float rsum16(float v) {
  v += __shfl_xor(v, 1, 32);
  v += __shfl_xor(v, 2, 32);
  v += __shfl_xor(v, 4, 32);
  v += __shfl_xor(v, 8, 32);
  return v;
}

// ---------------------------------------------------------------------------
// TDM 2D tile load: global -> LDS (packed rows). Descriptor per ISA
// CDNA5 ISA (D# group0/group1). data_size = 2 bytes.
// group0: [1:0]=count=1, [63:32]=lds_addr, [120:64]=global_addr, [127:126]=2.
// group1: data_size(2B)=1 @bits17:16; tensor_dim0 @[79:48]; tensor_dim1
//         @[111:80]; tile_dim0 @[127:112]; tile_dim1 @[143:128];
//         tensor_dim0_stride @[207:160].
// Toolchain here exposes the 6-arg builtin: (v4u, v8i, v4i, v4i, v8i, cpol).
// ---------------------------------------------------------------------------
__device__ __forceinline__ void tdm_load_2d(unsigned lds_off, const __bf16* g,
                                            int tile0, int tile1, int tdim0,
                                            int tdim1, int stride0) {
  unsigned long long ga = (unsigned long long)(uintptr_t)g;
  v4u g0;
  g0.x = 1u;                                    // count = 1 valid descriptor
  g0.y = lds_off;                               // LDS byte address
  g0.z = (unsigned)ga;                          // global addr [95:64]
  g0.w = (unsigned)(ga >> 32) | (2u << 30);     // addr msbs + type=2 (image)
  v8i g1;
  g1[0] = 0x00010000;                           // data_size = 2 bytes
  g1[1] = (tdim0 & 0xFFFF) << 16;               // tensor_dim0[15:0]
  g1[2] = ((tdim0 >> 16) & 0xFFFF) | ((tdim1 & 0xFFFF) << 16);
  g1[3] = ((tdim1 >> 16) & 0xFFFF) | ((tile0 & 0xFFFF) << 16);
  g1[4] = tile1 & 0xFFFF;                       // tile_dim1 (tile_dim2 = 0)
  g1[5] = stride0;                              // tensor_dim0_stride[31:0]
  g1[6] = 0;
  g1[7] = 0;
  v4i z4 = {0, 0, 0, 0};
  v8i z8;
#pragma unroll
  for (int i = 0; i < 8; ++i) z8[i] = 0;
  __builtin_amdgcn_tensor_load_to_lds(g0, g1, z4, z4, z8, 0);
}

// ---------------------------------------------------------------------------
// fp32 -> bf16 conversion (grid-stride)
// ---------------------------------------------------------------------------
__global__ void f2bf_kernel(const void* inv, void* outv, long n) {
  const float* in = (const float*)inv;
  __bf16* out = (__bf16*)outv;
  long i = (long)blockIdx.x * blockDim.x + threadIdx.x;
  long stride = (long)gridDim.x * blockDim.x;
  for (; i < n; i += stride) out[i] = (__bf16)in[i];
}

// ---------------------------------------------------------------------------
// bf16 GEMM: C[M,N] = A[M,K] * W[N,K]^T, fp32 accumulate.
// One wave computes a 32x64 C block (2 m-subtiles x 4 n-tiles -> 8 WMMAs per
// 32-k chunk), with software-pipelined (double-buffered) fragment loads so
// next-chunk global loads overlap current WMMAs.
// Epilogue modes:
//   0: fp32 row-major out0[m*N + n]
//   1: bf16 q buffer  out0[((b*NH+h)*S + s)*QHD + d]   (n = h*QHD + d)
//   2: bf16 split:    r = n&255, h = n>>8;
//      r<128 -> key out0[((b*NH+h)*S+s)*QHD + r]
//      else  -> v^T out1[((b*NH+h)*DV + r-128)*S + s]
// ---------------------------------------------------------------------------
#define GEMM_WPB 8
__global__ __launch_bounds__(256) void gemm_bf16_kernel(
    const void* Av, const void* Wv, void* out0v, void* out1v, int M, int N,
    int K, int mode) {
  const __bf16* A = (const __bf16*)Av;
  const __bf16* W = (const __bf16*)Wv;
  int lane = threadIdx.x & 31;
  int wv = threadIdx.x >> 5;
  int l16 = lane & 15, hi = lane >> 4;
  int ntiles = N >> 6;
  int tile = blockIdx.x * GEMM_WPB + wv;
  int mt = tile / ntiles, nt = tile % ntiles;
  int m0 = mt << 5, n0 = nt << 6;

  const __bf16* ar0 = A + (size_t)(m0 + l16) * K;
  const __bf16* ar1 = ar0 + (size_t)16 * K;
  const __bf16* wr = W + (size_t)(n0 + l16) * K;

  f32x8 acc[2][4];
#pragma unroll
  for (int s = 0; s < 2; ++s)
#pragma unroll
    for (int t = 0; t < 4; ++t)
#pragma unroll
      for (int i = 0; i < 8; ++i) acc[s][t][i] = 0.0f;

  bf16x16 a0 = lda(ar0, 0, hi);
  bf16x16 a1 = lda(ar1, 0, hi);
  bf16x16 b[4];
#pragma unroll
  for (int t = 0; t < 4; ++t) b[t] = ldb(wr + (size_t)t * 16 * K, 0, hi);

  for (int kb = 0; kb + 32 < K; kb += 32) {
    int kn = kb + 32;
    __builtin_prefetch(ar0 + kn + 32, 0, 1);  // global_prefetch_b8
    __builtin_prefetch(wr + kn + 32, 0, 1);
    // start next-chunk loads before consuming current fragments
    bf16x16 na0 = lda(ar0, kn, hi);
    bf16x16 na1 = lda(ar1, kn, hi);
    bf16x16 nb[4];
#pragma unroll
    for (int t = 0; t < 4; ++t) nb[t] = ldb(wr + (size_t)t * 16 * K, kn, hi);
#pragma unroll
    for (int t = 0; t < 4; ++t) {
      acc[0][t] = wmma_bf(a0, b[t], acc[0][t]);
      acc[1][t] = wmma_bf(a1, b[t], acc[1][t]);
    }
    a0 = na0;
    a1 = na1;
#pragma unroll
    for (int t = 0; t < 4; ++t) b[t] = nb[t];
  }
#pragma unroll
  for (int t = 0; t < 4; ++t) {
    acc[0][t] = wmma_bf(a0, b[t], acc[0][t]);
    acc[1][t] = wmma_bf(a1, b[t], acc[1][t]);
  }

  if (mode == 0) {
    float* o = (float*)out0v;
#pragma unroll
    for (int ms = 0; ms < 2; ++ms)
#pragma unroll
      for (int t = 0; t < 4; ++t)
#pragma unroll
        for (int i = 0; i < 8; ++i) {
          int m = m0 + ms * 16 + i + 8 * hi;
          int n = n0 + t * 16 + l16;
          o[(size_t)m * N + n] = acc[ms][t][i];
        }
  } else if (mode == 1) {
    __bf16* o = (__bf16*)out0v;
#pragma unroll
    for (int ms = 0; ms < 2; ++ms)
#pragma unroll
      for (int t = 0; t < 4; ++t)
#pragma unroll
        for (int i = 0; i < 8; ++i) {
          int m = m0 + ms * 16 + i + 8 * hi;
          int n = n0 + t * 16 + l16;
          int bb = m >> 11, s = m & (S_ - 1);
          int h = n / QHD_, d = n % QHD_;
          o[((size_t)(bb * NH_ + h) * S_ + s) * QHD_ + d] =
              (__bf16)acc[ms][t][i];
        }
  } else {
    __bf16* key = (__bf16*)out0v;
    __bf16* vt = (__bf16*)out1v;
#pragma unroll
    for (int ms = 0; ms < 2; ++ms)
#pragma unroll
      for (int t = 0; t < 4; ++t)
#pragma unroll
        for (int i = 0; i < 8; ++i) {
          int m = m0 + ms * 16 + i + 8 * hi;
          int n = n0 + t * 16 + l16;
          int bb = m >> 11, s = m & (S_ - 1);
          int h = n >> 8, r = n & 255;
          if (r < DQK_)
            key[((size_t)(bb * NH_ + h) * S_ + s) * QHD_ + r] =
                (__bf16)acc[ms][t][i];
          else
            vt[((size_t)(bb * NH_ + h) * DV_ + (r - DQK_)) * S_ + s] =
                (__bf16)acc[ms][t][i];
        }
  }
}

// ---------------------------------------------------------------------------
// LayerNorm over latent 512 + RoPE on the 64 shared k_pe dims (broadcast to
// all 16 heads of the key buffer). One 256-thread block per token row.
// ---------------------------------------------------------------------------
__global__ __launch_bounds__(256) void ln_rope_kernel(const void* kvcv,
                                                      const float* g,
                                                      const float* bta,
                                                      void* kvlnv, void* keyv) {
  const float* kvc = (const float*)kvcv;
  __bf16* kvln = (__bf16*)kvlnv;
  __bf16* key = (__bf16*)keyv;
  int m = blockIdx.x;
  int t = threadIdx.x;
  const float* row = kvc + (size_t)m * (LAT_ + DPE_);
  float x0 = row[t], x1 = row[t + 256];
  __shared__ float s1[256], s2[256];
  s1[t] = x0 + x1;
  s2[t] = x0 * x0 + x1 * x1;
  __syncthreads();
  for (int off = 128; off > 0; off >>= 1) {
    if (t < off) {
      s1[t] += s1[t + off];
      s2[t] += s2[t + off];
    }
    __syncthreads();
  }
  float mu = s1[0] * (1.0f / LAT_);
  float var = s2[0] * (1.0f / LAT_) - mu * mu;
  float rstd = rsqrtf(var + 1e-5f);
  kvln[(size_t)m * LAT_ + t] = (__bf16)((x0 - mu) * rstd * g[t] + bta[t]);
  kvln[(size_t)m * LAT_ + t + 256] =
      (__bf16)((x1 - mu) * rstd * g[t + 256] + bta[t + 256]);

  if (t < 32) {
    int s = m & (S_ - 1);
    int b = m >> 11;
    float p1 = row[LAT_ + t], p2 = row[LAT_ + 32 + t];
    float inv = powf(10000.0f, -(float)t / 32.0f);
    float ang = (float)s * inv;
    float c = cosf(ang), sn = sinf(ang);
    __bf16 o1 = (__bf16)(p1 * c - p2 * sn);
    __bf16 o2 = (__bf16)(p2 * c + p1 * sn);
    for (int h = 0; h < NH_; ++h) {
      size_t base = ((size_t)(b * NH_ + h) * S_ + s) * QHD_;
      key[base + DQK_ + t] = o1;
      key[base + DQK_ + 32 + t] = o2;
    }
  }
}

// ---------------------------------------------------------------------------
// RoPE applied in-place to q_pe (dims 128..191) of the q buffer (b,h,s,192).
// ---------------------------------------------------------------------------
__global__ __launch_bounds__(32) void rope_q_kernel(void* qv) {
  __bf16* q = (__bf16*)qv;
  int row = blockIdx.x;  // (b*NH + h)*S + s
  int t = threadIdx.x;   // 0..31
  int s = row & (S_ - 1);
  __bf16* r = q + (size_t)row * QHD_;
  float x1 = (float)r[DQK_ + t];
  float x2 = (float)r[DQK_ + 32 + t];
  float inv = powf(10000.0f, -(float)t / 32.0f);
  float ang = (float)s * inv;
  float c = cosf(ang), sn = sinf(ang);
  r[DQK_ + t] = (__bf16)(x1 * c - x2 * sn);
  r[DQK_ + 32 + t] = (__bf16)(x2 * c + x1 * sn);
}

// ---------------------------------------------------------------------------
// Flash attention. 4 waves/block share one (b,h); each wave owns a 16-query
// tile. Wave 0 drives the TDM to double-buffer K (32x192) and V^T (128x32)
// chunks into LDS (s_wait_tensorcnt + block barrier handshakes). Compute:
// 12 score WMMAs + 8 P.V WMMAs per 32-key chunk, online softmax, P-tile
// C-layout -> A-layout transpose through per-wave LDS (s_wait_dscnt).
// q   : (B,NH,S,192) bf16    key : (B,NH,S,192) bf16
// vt  : (B,NH,128,S) bf16    ctx : (B,S,NH*128) bf16
// ---------------------------------------------------------------------------
__global__ __launch_bounds__(128) void mla_attn_kernel(const void* qv,
                                                       const void* keyv,
                                                       const void* vtv,
                                                       void* ctxv) {
  const __bf16* q = (const __bf16*)qv;
  const __bf16* key = (const __bf16*)keyv;
  const __bf16* vt = (const __bf16*)vtv;
  __bf16* ctx = (__bf16*)ctxv;

  int lane = threadIdx.x & 31, wv = threadIdx.x >> 5;
  int l16 = lane & 15, hi = lane >> 4;
  int wave = blockIdx.x * 4 + wv;  // 4096 waves total
  int qt = wave & 127;             // S/16
  int bh = wave >> 7;              // b*NH + h (same for all 4 waves in block)
  int b = bh >> 4, h = bh & 15;
  int bhB = blockIdx.x >> 5;       // block-uniform copy of bh for TDM

  __shared__ __align__(64) __bf16 kls[2][32 * QHD_];  // 2 x 12 KB
  __shared__ __align__(64) __bf16 vls[2][DV_ * 32];   // 2 x 8 KB
  __shared__ __align__(32) __bf16 psm[4][16 * 32];    // P transpose, 4 KB
  __bf16* pl = psm[wv];

  const __bf16* kglob = key + (size_t)bhB * S_ * QHD_;
  const __bf16* vglob = vt + (size_t)bhB * DV_ * S_;

  // Q fragments for all 6 d-chunks (resident: 48 VGPRs).
  const __bf16* qrow = q + ((size_t)bh * S_ + qt * 16 + l16) * QHD_;
  bf16x16 qf[6];
#pragma unroll
  for (int dc = 0; dc < 6; ++dc) qf[dc] = lda(qrow, dc * 32, hi);

  f32x8 vacc[8];
#pragma unroll
  for (int t = 0; t < 8; ++t)
#pragma unroll
    for (int i = 0; i < 8; ++i) vacc[t][i] = 0.0f;
  float mrow[8], rsum[8];
#pragma unroll
  for (int i = 0; i < 8; ++i) {
    mrow[i] = -3.0e38f;
    rsum[i] = 0.0f;
  }

  const float SC = 0.07216878364870323f;  // 1/sqrt(192)
  const int NIT = S_ / 32;                // 64 chunks

  // TDM prologue: preload chunks 0 and 1 into both buffers (wave 0 only).
  if (wv == 0) {
#pragma unroll
    for (int p = 0; p < 2; ++p) {
      tdm_load_2d((unsigned)(uintptr_t)&kls[p][0], kglob + (size_t)p * 32 * QHD_,
                  QHD_, 32, QHD_, S_, QHD_);
      tdm_load_2d((unsigned)(uintptr_t)&vls[p][0], vglob + p * 32, 32, DV_, S_,
                  DV_, S_);
    }
  }

  for (int it = 0; it < NIT; ++it) {
    int kb = it * 32;
    int cur = it & 1;
    if (wv == 0) {
      if (it + 2 < NIT)
        __builtin_amdgcn_s_wait_tensorcnt(2);  // chunk `it` landed
      else
        __builtin_amdgcn_s_wait_tensorcnt(0);  // drain
    }
    __syncthreads();  // publish LDS tile to all waves

    // ---- score tiles: 16 queries x 32 keys (K from LDS) ------------------
    f32x8 s0, s1;
#pragma unroll
    for (int i = 0; i < 8; ++i) {
      s0[i] = 0.0f;
      s1[i] = 0.0f;
    }
#pragma unroll
    for (int dc = 0; dc < 6; ++dc) {
      bf16x16 k0 = *(const bf16x16*)(&kls[cur][l16 * QHD_ + dc * 32 + 16 * hi]);
      s0 = wmma_bf(qf[dc], k0, s0);
      bf16x16 k1 =
          *(const bf16x16*)(&kls[cur][(16 + l16) * QHD_ + dc * 32 + 16 * hi]);
      s1 = wmma_bf(qf[dc], k1, s1);
    }
    // ---- online softmax update ------------------------------------------
#pragma unroll
    for (int i = 0; i < 8; ++i) {
      float x0 = s0[i] * SC, x1 = s1[i] * SC;
      float tm = rmax16(fmaxf(x0, x1));
      float mnew = fmaxf(mrow[i], tm);
      float alpha = __expf(mrow[i] - mnew);
      float p0 = __expf(x0 - mnew);
      float p1 = __expf(x1 - mnew);
      float ts = rsum16(p0 + p1);
      rsum[i] = rsum[i] * alpha + ts;
      mrow[i] = mnew;
#pragma unroll
      for (int t = 0; t < 8; ++t) vacc[t][i] *= alpha;
      int r = i + 8 * hi;  // C-layout row
      pl[r * 32 + l16] = (__bf16)p0;
      pl[r * 32 + 16 + l16] = (__bf16)p1;
    }
    // ---- C-layout -> A-layout transpose through LDS ----------------------
    asm volatile("s_wait_dscnt 0x0" ::: "memory");
    bf16x16 pf = cat8(*(const bf16x8*)(pl + l16 * 32 + 16 * hi),
                      *(const bf16x8*)(pl + l16 * 32 + 16 * hi + 8));
    // ---- P (16x32) . V (32x128) (V from LDS) -----------------------------
#pragma unroll
    for (int t = 0; t < 8; ++t) {
      bf16x16 vfr = *(const bf16x16*)(&vls[cur][(t * 16 + l16) * 32 + 16 * hi]);
      vacc[t] = wmma_bf(pf, vfr, vacc[t]);
    }
    __syncthreads();  // all waves done with buffer `cur`
    if (wv == 0 && it + 2 < NIT) {
      int kn = kb + 64;
      tdm_load_2d((unsigned)(uintptr_t)&kls[cur][0],
                  kglob + (size_t)kn * QHD_, QHD_, 32, QHD_, S_, QHD_);
      tdm_load_2d((unsigned)(uintptr_t)&vls[cur][0], vglob + kn, 32, DV_, S_,
                  DV_, S_);
    }
  }

  // ---- finalize + store ctx ---------------------------------------------
#pragma unroll
  for (int i = 0; i < 8; ++i) {
    float inv = 1.0f / rsum[i];
#pragma unroll
    for (int t = 0; t < 8; ++t) vacc[t][i] *= inv;
  }
#pragma unroll
  for (int t = 0; t < 8; ++t)
#pragma unroll
    for (int i = 0; i < 8; ++i) {
      int s = qt * 16 + i + 8 * hi;
      ctx[((size_t)b * S_ + s) * (NH_ * DV_) + h * DV_ + t * 16 + l16] =
          (__bf16)vacc[t][i];
    }
}

// ---------------------------------------------------------------------------
// Host-side orchestration
// ---------------------------------------------------------------------------
extern "C" void kernel_launch(void* const* d_in, const int* in_sizes, int n_in,
                              void* d_out, int out_size, void* d_ws,
                              size_t ws_size, hipStream_t stream) {
  (void)in_sizes;
  (void)n_in;
  (void)out_size;
  (void)ws_size;
  const float* hs = (const float*)d_in[0];
  const float* qw = (const float*)d_in[1];
  const float* kdw = (const float*)d_in[2];
  const float* lng = (const float*)d_in[3];
  const float* lnb = (const float*)d_in[4];
  const float* kuw = (const float*)d_in[5];
  const float* ow = (const float*)d_in[6];

  const long nHS = (long)B_ * S_ * H_;
  const long nQW = (long)NH_ * QHD_ * H_;
  const long nKDW = (long)(LAT_ + DPE_) * H_;
  const long nKUW = (long)NH_ * 256 * LAT_;
  const long nOW = (long)H_ * NH_ * DV_;
  const long nQB = (long)B_ * NH_ * S_ * QHD_;
  const long nKVC = (long)B_ * S_ * (LAT_ + DPE_);
  const long nLN = (long)B_ * S_ * LAT_;
  const long nVT = (long)B_ * NH_ * DV_ * S_;
  const long nCTX = (long)B_ * S_ * NH_ * DV_;

  char* w = (char*)d_ws;
  auto alloc = [&](size_t bytes) -> void* {
    void* p = (void*)w;
    w += (bytes + 255) & ~(size_t)255;
    return p;
  };
  void* hs_bf = alloc(nHS * 2);
  void* qw_bf = alloc(nQW * 2);
  void* kdw_bf = alloc(nKDW * 2);
  void* kuw_bf = alloc(nKUW * 2);
  void* ow_bf = alloc(nOW * 2);
  void* qbuf = alloc(nQB * 2);   // (B,NH,S,192) bf16
  void* kvc = alloc(nKVC * 4);   // fp32
  void* kvln = alloc(nLN * 2);   // (B*S,512) bf16
  void* keyb = alloc(nQB * 2);   // (B,NH,S,192) bf16
  void* vtb = alloc(nVT * 2);    // (B,NH,128,S) bf16
  void* ctxb = alloc(nCTX * 2);  // (B,S,2048) bf16

  // 1. fp32 -> bf16 conversions
  f2bf_kernel<<<4096, 256, 0, stream>>>(hs, hs_bf, nHS);
  f2bf_kernel<<<4096, 256, 0, stream>>>(qw, qw_bf, nQW);
  f2bf_kernel<<<2048, 256, 0, stream>>>(kdw, kdw_bf, nKDW);
  f2bf_kernel<<<2048, 256, 0, stream>>>(kuw, kuw_bf, nKUW);
  f2bf_kernel<<<4096, 256, 0, stream>>>(ow, ow_bf, nOW);

  const int Mrows = B_ * S_;  // 4096 -> 128 m-tiles of 32
  // 2. q = hs @ q_w^T  -> qbuf (mode 1)
  {
    int N = NH_ * QHD_;  // 3072
    int tiles = (Mrows / 32) * (N / 64);
    gemm_bf16_kernel<<<tiles / GEMM_WPB, 256, 0, stream>>>(
        hs_bf, qw_bf, qbuf, nullptr, Mrows, N, H_, 1);
  }
  // 3. kvc = hs @ kv_down_w^T -> fp32 (mode 0)
  {
    int N = LAT_ + DPE_;  // 576
    int tiles = (Mrows / 32) * (N / 64);
    gemm_bf16_kernel<<<tiles / GEMM_WPB, 256, 0, stream>>>(
        hs_bf, kdw_bf, kvc, nullptr, Mrows, N, H_, 0);
  }
  // 4. LayerNorm + k_pe RoPE broadcast into key buffer
  ln_rope_kernel<<<Mrows, 256, 0, stream>>>(kvc, lng, lnb, kvln, keyb);
  // 5. kv = kv_ln @ kv_up_w^T -> k_nope into keyb, v into vtb transposed
  {
    int N = NH_ * 256;  // 4096
    int tiles = (Mrows / 32) * (N / 64);
    gemm_bf16_kernel<<<tiles / GEMM_WPB, 256, 0, stream>>>(
        kvln, kuw_bf, keyb, vtb, Mrows, N, LAT_, 2);
  }
  // 6. RoPE on q_pe (in-place in qbuf)
  rope_q_kernel<<<B_ * NH_ * S_, 32, 0, stream>>>(qbuf);
  // 7. attention -> ctx (TDM-fed)
  mla_attn_kernel<<<(B_ * NH_ * (S_ / 16)) / 4, 128, 0, stream>>>(qbuf, keyb,
                                                                  vtb, ctxb);
  // 8. out = ctx @ out_w^T -> d_out fp32 (mode 0)
  {
    int N = H_;  // 2048
    int tiles = (Mrows / 32) * (N / 64);
    gemm_bf16_kernel<<<tiles / GEMM_WPB, 256, 0, stream>>>(
        ctxb, ow_bf, d_out, nullptr, Mrows, N, NH_ * DV_, 0);
  }
}
